// CLSAEncoder_15410342658408
// MI455X (gfx1250) — matrix operand order repaired and checked by reading, not compiled, not run on hardware
//
#include <hip/hip_runtime.h>
#include <math.h>

// Problem constants (match reference)
#define ROWS 32
#define COLS 6
#define CH   16
#define KW   3
#define T    256
#define BB   16          // batch
#define D    3072        // ROWS*COLS*CH
#define D2   6144        // 2*D
#define NEGV (-1e9f)

typedef __attribute__((ext_vector_type(16))) __bf16       v16bf;
typedef __attribute__((ext_vector_type(8)))  float        v8f;
typedef __attribute__((ext_vector_type(8)))  unsigned int v8u;
typedef __attribute__((ext_vector_type(4)))  unsigned int u32x4;
typedef __attribute__((ext_vector_type(8)))  unsigned int u32x8;

union Frag { v8u u; v16bf b; };

__device__ __forceinline__ unsigned short f2bf(float x) {
    union { float f; unsigned int u; } c; c.f = x;
    unsigned int r = c.u + 0x7fffu + ((c.u >> 16) & 1u);  // round-to-nearest-even
    return (unsigned short)(r >> 16);
}

// ---------------- fp32 -> bf16 weight conversion (one-time) ----------------
__global__ void cvt_bf16_kernel(const float* __restrict__ src,
                                unsigned short* __restrict__ dst, int n) {
    for (int i = blockIdx.x * blockDim.x + threadIdx.x; i < n;
         i += gridDim.x * blockDim.x)
        dst[i] = f2bf(src[i]);
}

// ---------------- ConvLSTM cell step ----------------
// layer0: xin = x_flat (B,T,ROWS*COLS), cin=1 ; layer1: xin = ref0 (T,B,D), cin=CH
// h_prev read from raw[t-1] (zeros at t==0); c updated in-place per element.
__global__ void cell_kernel(const float* __restrict__ xin, int layer, int t,
                            const float* __restrict__ wx, const float* __restrict__ bx,
                            const float* __restrict__ wh, const float* __restrict__ bh,
                            float* __restrict__ raw,       // (T,B,D), writes raw[t]
                            float* __restrict__ cstate,    // (B,D)
                            unsigned short* __restrict__ cat) { // (B,D2) bf16
    int tid = blockIdx.x * blockDim.x + threadIdx.x;
    if (tid >= BB * D) return;
    int b   = tid / D;
    int rem = tid % D;            // (row*CH + ch)*COLS + col
    int col = rem % COLS;
    int rc  = rem / COLS;
    int ch  = rc % CH;
    int row = rc / CH;
    int cin = (layer == 0) ? 1 : CH;

    const float* hprev = raw + (size_t)(t - 1) * BB * D + (size_t)b * D;

    float g[4];
    #pragma unroll
    for (int gi = 0; gi < 4; ++gi) {
        int oc = gi * CH + ch;
        float acc = bx[oc] + bh[oc];
        for (int ci = 0; ci < cin; ++ci) {
            for (int dk = 0; dk < KW; ++dk) {
                int cc = col + dk - 1;
                if (cc < 0 || cc >= COLS) continue;
                float xv;
                if (layer == 0)
                    xv = xin[((size_t)b * T + t) * (ROWS * COLS) + row * COLS + cc];
                else
                    xv = xin[(size_t)t * BB * D + (size_t)b * D + (row * CH + ci) * COLS + cc];
                acc += wx[(oc * cin + ci) * KW + dk] * xv;
            }
        }
        if (t > 0) {
            for (int ci = 0; ci < CH; ++ci)
                for (int dk = 0; dk < KW; ++dk) {
                    int cc = col + dk - 1;
                    if (cc < 0 || cc >= COLS) continue;
                    acc += wh[(oc * CH + ci) * KW + dk] * hprev[(row * CH + ci) * COLS + cc];
                }
        }
        g[gi] = acc;
    }
    float c_prev = (t > 0) ? cstate[tid] : 0.f;
    float ig = 1.f / (1.f + __expf(-g[0]));
    float fg = 1.f / (1.f + __expf(-g[1]));
    float og = 1.f / (1.f + __expf(-g[2]));
    float gg = tanhf(g[3]);
    float cn = fg * c_prev + ig * gg;
    float hn = og * tanhf(cn);
    cstate[tid] = cn;
    raw[(size_t)t * BB * D + tid] = hn;
    cat[(size_t)b * D2 + rem] = f2bf(hn);   // first half of concat
}

// ---------------- attention scores: score[b][tp] = ref[tp,b,:] . h[b,:] ----------------
__global__ void scores_kernel(const float* __restrict__ ref, long long st, long long sb,
                              const float* __restrict__ raw, int t,
                              float* __restrict__ scores) {
    int blk = blockIdx.x;            // b*T + tp
    int b = blk / T, tp = blk % T;
    if (tp >= t) return;
    const float* rv = ref + (size_t)tp * st + (size_t)b * sb;
    const float* hv = raw + (size_t)t * BB * D + (size_t)b * D;
    __shared__ float red[128];
    float s = 0.f;
    for (int d = threadIdx.x; d < D; d += 128) s += rv[d] * hv[d];
    red[threadIdx.x] = s; __syncthreads();
    for (int off = 64; off; off >>= 1) {
        if (threadIdx.x < off) red[threadIdx.x] += red[threadIdx.x + off];
        __syncthreads();
    }
    if (threadIdx.x == 0) scores[blk] = red[0];
}

// ---------------- masked softmax; writes w buffer + attention output row ----------------
__global__ void softmax_kernel(const float* __restrict__ scores, int t,
                               float* __restrict__ w, float* __restrict__ aout) {
    int b = blockIdx.x, tp = threadIdx.x;  // 256 threads
    float s = (tp < t) ? scores[b * T + tp] : NEGV;
    __shared__ float red[256];
    red[tp] = s; __syncthreads();
    for (int off = 128; off; off >>= 1) {
        if (tp < off) red[tp] = fmaxf(red[tp], red[tp + off]);
        __syncthreads();
    }
    float m = red[0]; __syncthreads();
    float e = (tp < t) ? __expf(s - m) : 0.f;
    red[tp] = e; __syncthreads();
    for (int off = 128; off; off >>= 1) {
        if (tp < off) red[tp] += red[tp + off];
        __syncthreads();
    }
    float denom = red[0];
    float wv = (t > 0 && tp < t) ? (e / denom) : 0.f;
    w[b * T + tp] = wv;
    aout[(size_t)b * T * T + (size_t)t * T + tp] = wv;
}

// ---------------- context: cat[b][D+d] = bf16( sum_{tp<t} w[b][tp]*raw[tp,b,d] ) -------
__global__ void ctx_kernel(const float* __restrict__ w, const float* __restrict__ raw,
                           int t, unsigned short* __restrict__ cat) {
    int tid = blockIdx.x * blockDim.x + threadIdx.x;
    if (tid >= BB * D) return;
    int b = tid / D, d = tid % D;
    float acc = 0.f;
    for (int tp = 0; tp < t; ++tp)
        acc += w[b * T + tp] * raw[(size_t)tp * BB * D + (size_t)b * D + d];
    cat[(size_t)b * D2 + D + d] = f2bf(acc);
}

// ---------------- refine GEMM: dst = tanh(cat(16xD2,bf16) @ WfT + bias) via WMMA -------
// 192 N-tiles; 24 blocks x 4 waves x 2 tiles/wave. A staged into double-buffered LDS by
// the Tensor Data Mover (tensor_load_to_lds), overlapped with compute; weights (bf16,
// L2-resident) read directly, software-pipelined into back-to-back WMMAs.
#define CHUNK_K   768
#define CHUNK_DW  (CHUNK_K / 2)        // 384 dwords per row
#define NCHUNK    (D2 / CHUNK_K)       // 8
#define STEPS     (CHUNK_K / 32)       // 24 wmma steps per chunk

// Issue a TDM 2-D tile copy: tile 16 rows x CHUNK_K bf16 from cat (row stride D2) -> LDS.
__device__ __forceinline__ void tdm_load_chunk(const unsigned short* gsrc,
                                               unsigned lds_off) {
    unsigned long long ga = (unsigned long long)(size_t)gsrc;
    u32x4 g0;
    g0.x = 1u;                                          // count=1, user descriptor
    g0.y = lds_off;                                     // LDS byte address
    g0.z = (unsigned)(ga & 0xffffffffu);                // global_addr[31:0]
    g0.w = (unsigned)((ga >> 32) & 0x01ffffffu) | (2u << 30); // global_addr[56:32] | type=2
    u32x8 g1;
    g1.s0 = (1u << 16);                                 // wg_mask=0, data_size=1 (2 bytes)
    g1.s1 = ((unsigned)D2 & 0xffffu) << 16;             // tensor_dim0[15:0]
    g1.s2 = ((unsigned)D2 >> 16) | (16u << 16);         // tensor_dim0[31:16] | tensor_dim1 lo
    g1.s3 = ((unsigned)CHUNK_K) << 16;                  // tensor_dim1 hi | tile_dim0
    g1.s4 = 16u;                                        // tile_dim1=16, tile_dim2=0
    g1.s5 = (unsigned)D2;                               // tensor_dim0_stride[31:0]
    g1.s6 = 0u;                                         // stride hi / dim1_stride lo
    g1.s7 = 0u;
    asm volatile("tensor_load_to_lds %0, %1" :: "s"(g0), "s"(g1) : "memory");
}

__device__ __forceinline__ void loadA(Frag& a, const unsigned int* ldsbuf,
                                      int mrow, int hf, int s) {
    int base = mrow * CHUNK_DW + 16 * s + 8 * hf;
    #pragma unroll
    for (int i = 0; i < 4; ++i) {
        a.u[i]     = ldsbuf[base + i];
        a.u[4 + i] = ldsbuf[base + 4 + i];
    }
}
__device__ __forceinline__ void loadW(Frag& w, const unsigned int* wrow, int s) {
    #pragma unroll
    for (int i = 0; i < 8; ++i) w.u[i] = wrow[16 * s + i];
}

__global__ void __launch_bounds__(128)
refine_kernel(const unsigned short* __restrict__ catb,   // (16, D2) bf16
              const unsigned int* __restrict__ Wb,       // bf16 weights as dwords (row = D2/2)
              const float* __restrict__ bias, int t,
              float* __restrict__ dst, long long dt, long long dbb) {
    __shared__ unsigned int lds[2][16 * CHUNK_DW];       // 2 x 24KB double buffer
    int lane = threadIdx.x & 31;
    int wave = threadIdx.x >> 5;
    int tb   = (blockIdx.x * 4 + wave) * 2;              // first of 2 N-tiles for this wave
    int mrow = lane & 15;
    int hf   = lane >> 4;

    // Kick off TDM for chunk 0 (wave 0 only; TENSORcnt is per-wave).
    if (threadIdx.x < 32)
        tdm_load_chunk(catb, (unsigned)(size_t)(void*)&lds[0][0]);

    v8f acc0 = {}, acc1 = {};
    for (int c = 0; c < NCHUNK; ++c) {
        int kc = c * CHUNK_K;
        if (threadIdx.x < 32) __builtin_amdgcn_s_wait_tensorcnt(0);
        __syncthreads();                                  // publish LDS tile to all waves
        // Prefetch next chunk's A tile via TDM into the other buffer (overlaps compute).
        if (c + 1 < NCHUNK && threadIdx.x < 32)
            tdm_load_chunk(catb + (size_t)(kc + CHUNK_K),
                           (unsigned)(size_t)(void*)&lds[(c + 1) & 1][0]);

        const unsigned int* ldsbuf = &lds[c & 1][0];
        const unsigned int* wr0 = Wb + (size_t)((tb + 0) * 16 + mrow) * (D2 / 2) + (kc >> 1) + 8 * hf;
        const unsigned int* wr1 = Wb + (size_t)((tb + 1) * 16 + mrow) * (D2 / 2) + (kc >> 1) + 8 * hf;
        if (c + 1 < NCHUNK) {                             // warm near cache for next chunk
            __builtin_prefetch((const void*)(wr0 + CHUNK_DW), 0, 0);
            __builtin_prefetch((const void*)(wr1 + CHUNK_DW), 0, 0);
        }

        // Software-pipelined inner loop: load frags for s+1 while WMMAs consume s.
        Frag a, w0, w1;
        loadA(a, ldsbuf, mrow, hf, 0);
        loadW(w0, wr0, 0);
        loadW(w1, wr1, 0);
        #pragma unroll
        for (int s = 0; s < STEPS; ++s) {
            Frag an, w0n, w1n;
            int sn = (s + 1 < STEPS) ? s + 1 : s;
            loadA(an, ldsbuf, mrow, hf, sn);
            loadW(w0n, wr0, sn);
            loadW(w1n, wr1, sn);
            acc0 = __builtin_amdgcn_wmma_f32_16x16x32_bf16(
                false, a.b, false, w0.b, (short)0, acc0, false, false);
            acc1 = __builtin_amdgcn_wmma_f32_16x16x32_bf16(
                false, a.b, false, w1.b, (short)0, acc1, false, false);
            a = an; w0 = w0n; w1 = w1n;
        }
        __syncthreads();                                  // all waves done with this buffer
    }

    // Epilogue: C/D layout — VGPR r -> batch row M = r + 8*hf, column N = lane&15.
    #pragma unroll
    for (int nt = 0; nt < 2; ++nt) {
        int j = (tb + nt) * 16 + mrow;
        float bj = bias[j];
        #pragma unroll
        for (int r = 0; r < 8; ++r) {
            int brow = r + 8 * hf;
            float v = (nt == 0) ? acc0[r] : acc1[r];
            dst[(size_t)t * dt + (size_t)brow * dbb + j] = tanhf(v + bj);
        }
    }
}

extern "C" void kernel_launch(void* const* d_in, const int* in_sizes, int n_in,
                              void* d_out, int out_size, void* d_ws, size_t ws_size,
                              hipStream_t stream) {
    const float* x_flat = (const float*)d_in[0];
    const float* wx0 = (const float*)d_in[1];
    const float* bx0 = (const float*)d_in[2];
    const float* wh0 = (const float*)d_in[3];
    const float* bh0 = (const float*)d_in[4];
    const float* Wf0 = (const float*)d_in[5];
    const float* bf0 = (const float*)d_in[6];
    const float* wx1 = (const float*)d_in[7];
    const float* bx1 = (const float*)d_in[8];
    const float* wh1 = (const float*)d_in[9];
    const float* bh1 = (const float*)d_in[10];
    const float* Wf1 = (const float*)d_in[11];
    const float* bf1 = (const float*)d_in[12];

    float* enc = (float*)d_out;                    // (B,T,D)
    float* a0  = enc + (size_t)BB * T * D;         // (B,T,T)
    float* a1  = a0 + (size_t)BB * T * T;          // (B,T,T)

    char* ws = (char*)d_ws;
    size_t off = 0;
    auto alloc = [&](size_t bytes) -> void* {
        void* p = ws + off;
        off += (bytes + 255) & ~(size_t)255;
        return p;
    };
    unsigned short* WfB0 = (unsigned short*)alloc((size_t)D * D2 * 2);  // 37.75 MB
    unsigned short* WfB1 = (unsigned short*)alloc((size_t)D * D2 * 2);  // 37.75 MB
    float* raw0 = (float*)alloc((size_t)T * BB * D * 4);                // 50.3 MB
    float* ref0 = (float*)alloc((size_t)T * BB * D * 4);                // 50.3 MB
    float* raw1 = (float*)alloc((size_t)T * BB * D * 4);                // 50.3 MB
    float* c0   = (float*)alloc((size_t)BB * D * 4);
    float* c1   = (float*)alloc((size_t)BB * D * 4);
    float* scores = (float*)alloc((size_t)BB * T * 4);
    float* wbuf   = (float*)alloc((size_t)BB * T * 4);
    unsigned short* cat = (unsigned short*)alloc((size_t)BB * D2 * 2);
    (void)ws_size; (void)in_sizes; (void)n_in; (void)out_size;

    // One-time: bf16 weight copies (both stay L2-resident: 75.5 MB < 192 MB L2)
    const int nW = D * D2;
    cvt_bf16_kernel<<<2048, 256, 0, stream>>>(Wf0, WfB0, nW);
    cvt_bf16_kernel<<<2048, 256, 0, stream>>>(Wf1, WfB1, nW);

    const int nBD = BB * D;
    const dim3 cellGrid((nBD + 127) / 128);

    for (int t = 0; t < T; ++t) {
        // -------- layer 0 --------
        cell_kernel<<<cellGrid, 128, 0, stream>>>(x_flat, 0, t, wx0, bx0, wh0, bh0,
                                                  raw0, c0, cat);
        scores_kernel<<<BB * T, 128, 0, stream>>>(ref0, (long long)BB * D, (long long)D,
                                                  raw0, t, scores);
        softmax_kernel<<<BB, 256, 0, stream>>>(scores, t, wbuf, a0);
        ctx_kernel<<<cellGrid, 128, 0, stream>>>(wbuf, raw0, t, cat);
        refine_kernel<<<24, 128, 0, stream>>>(cat, (const unsigned int*)WfB0, bf0, t,
                                              ref0, (long long)BB * D, (long long)D);
        // -------- layer 1 (input = refined layer-0 state) --------
        cell_kernel<<<cellGrid, 128, 0, stream>>>(ref0, 1, t, wx1, bx1, wh1, bh1,
                                                  raw1, c1, cat);
        // ref1 history lives directly in enc output: elem = enc[b*T*D + tp*D + d]
        scores_kernel<<<BB * T, 128, 0, stream>>>(enc, (long long)D, (long long)T * D,
                                                  raw1, t, scores);
        softmax_kernel<<<BB, 256, 0, stream>>>(scores, t, wbuf, a1);
        ctx_kernel<<<cellGrid, 128, 0, stream>>>(wbuf, raw1, t, cat);
        refine_kernel<<<24, 128, 0, stream>>>(cat, (const unsigned int*)WfB1, bf1, t,
                                              enc, (long long)D, (long long)T * D);
    }
}
